// FGA_S_22479858828041
// MI455X (gfx1250) — matrix operand order repaired and checked by reading, not compiled, not run on hardware
//
#include <hip/hip_runtime.h>
#include <hip/hip_bf16.h>
#include <math.h>

// ---------------- problem constants ----------------
#define BN     2
#define CN     64
#define HN     128
#define WN     128
#define HWN    16384      // HN*WN
#define FACTOR 8
#define HWG    2048       // HWN / FACTOR
#define QKVC   320        // 5*CN

typedef _Float16 half16 __attribute__((ext_vector_type(16)));
typedef _Float16 half8  __attribute__((ext_vector_type(8)));
typedef float    float8 __attribute__((ext_vector_type(8)));

// ---------------- WMMA helpers (CDNA5, wave32) ----------------
static __device__ __forceinline__ float8 wmma_f32_16x16x32_f16(half16 a, half16 b, float8 c) {
  // 8 args: (neg_a, A, neg_b, B, c_mod, C, reuse_a, reuse_b)
  return __builtin_amdgcn_wmma_f32_16x16x32_f16(false, a, false, b, (short)0, c, false, false);
}

// load 16 halfs as two 8-half (16B) vectors into one fragment register set
static __device__ __forceinline__ half16 ld_a2(const _Float16* p0, const _Float16* p1) {
  half8 x = *(const half8*)p0;
  half8 y = *(const half8*)p1;
  half16 r;
#pragma unroll
  for (int i = 0; i < 8; ++i) { r[i] = x[i]; r[8 + i] = y[i]; }
  return r;
}

// ---------------- 1x1 qkv conv fused with feats*idx ----------------
__global__ void k_qkv_pw(const float* __restrict__ feats, const float* __restrict__ idx,
                         const float* __restrict__ wq, float* __restrict__ out) {
  int t = blockIdx.x * blockDim.x + threadIdx.x;
  if (t >= BN * QKVC * HWN) return;
  int p  = t % HWN;
  int oc = (t / HWN) % QKVC;
  int b  = t / (HWN * QKVC);
  const float* xb   = feats + (size_t)b * CN * HWN + p;
  const float* wrow = wq + oc * CN;  // (320,64,1,1)
  float s = 0.f;
#pragma unroll 8
  for (int ic = 0; ic < CN; ++ic) s = fmaf(wrow[ic], xb[(size_t)ic * HWN], s);
  out[t] = s * idx[b * HWN + p];     // conv(x*idx) == idx * conv(x) for 1x1
}

// ---------------- 3x3 depthwise conv, pad=1 ----------------
__global__ void k_dw(const float* __restrict__ in, const float* __restrict__ wd,
                     float* __restrict__ out) {
  int t = blockIdx.x * blockDim.x + threadIdx.x;
  if (t >= BN * QKVC * HWN) return;
  int p  = t % HWN;
  int ch = (t / HWN) % QKVC;
  int b  = t / (HWN * QKVC);
  int y = p >> 7, x = p & 127;
  const float* base = in + (size_t)(b * QKVC + ch) * HWN;
  const float* wr   = wd + ch * 9;  // (320,1,3,3)
  float s = 0.f;
#pragma unroll
  for (int ky = 0; ky < 3; ++ky) {
    int yy = y + ky - 1;
    if (yy < 0 || yy >= HN) continue;
#pragma unroll
    for (int kx = 0; kx < 3; ++kx) {
      int xx = x + kx - 1;
      if (xx < 0 || xx >= WN) continue;
      s = fmaf(wr[ky * 3 + kx], base[yy * WN + xx], s);
    }
  }
  out[t] = s;
}

// ---------------- segmented bitonic argsort of idx ----------------
__global__ void k_sort_init(const float* __restrict__ idx, float* __restrict__ key,
                            int* __restrict__ val) {
  int t = blockIdx.x * blockDim.x + threadIdx.x;
  if (t >= BN * HWN) return;
  key[t] = idx[t];
  val[t] = t % HWN;
}

__global__ void k_bitonic(float* __restrict__ key, int* __restrict__ val, int k, int j) {
  int t = blockIdx.x * blockDim.x + threadIdx.x;
  if (t >= BN * HWN) return;
  int i = t % HWN, seg = t / HWN;
  int ixj = i ^ j;
  if (ixj <= i) return;
  int a = seg * HWN + i, c = seg * HWN + ixj;
  float ka = key[a], kc = key[c];
  int va = val[a], vc = val[c];
  bool gt = (ka > kc) || (ka == kc && va > vc);  // lexicographic => stable-equivalent
  bool up = ((i & k) == 0);
  if (gt == up) { key[a] = kc; key[c] = ka; val[a] = vc; val[c] = va; }
}

// ---------------- gather + l2norm + f16 pack ----------------
// Qn*/Kn*: (B*F, HWG, C) f16 row-major (c contiguous);  Vt: (B*F, C, HWG) f16
__global__ void k_gather(const float* __restrict__ qkv, const int* __restrict__ perm,
                         _Float16* __restrict__ Qn1, _Float16* __restrict__ Kn1,
                         _Float16* __restrict__ Qn2, _Float16* __restrict__ Kn2,
                         _Float16* __restrict__ Vt) {
  int t = blockIdx.x * blockDim.x + threadIdx.x;
  if (t >= BN * HWN) return;
  int p = t % HWN, b = t / HWN;
  int g = p >> 11, q = p & 2047;      // p = g*HWG + q  (reshape (c,factor,hw))
  int orig = perm[b * HWN + p];
  size_t rowOff = ((size_t)(b * FACTOR + g) * HWG + q) * CN;
  _Float16* outs[4] = {Qn1 + rowOff, Kn1 + rowOff, Qn2 + rowOff, Kn2 + rowOff};
#pragma unroll
  for (int s = 0; s < 4; ++s) {
    const float* src = qkv + ((size_t)(b * QKVC + s * CN) * HWN) + orig;
    float ss = 0.f;
    for (int ch = 0; ch < CN; ++ch) { float v = src[(size_t)ch * HWN]; ss = fmaf(v, v, ss); }
    float inv = 1.0f / fmaxf(sqrtf(ss), 1e-12f);
    for (int ch = 0; ch < CN; ++ch) outs[s][ch] = (_Float16)(src[(size_t)ch * HWN] * inv);
  }
  const float* srcv = qkv + ((size_t)(b * QKVC + 4 * CN) * HWN) + orig;
  _Float16* vt = Vt + (size_t)(b * FACTOR + g) * CN * HWG + q;
  for (int ch = 0; ch < CN; ++ch) vt[(size_t)ch * HWG] = (_Float16)srcv[(size_t)ch * HWN];
}

// ---------------- group means of V ----------------
__global__ void k_gmean(const _Float16* __restrict__ Vt, float* __restrict__ gm) {
  int bf = blockIdx.x;          // 16 blocks
  int c  = threadIdx.x;         // 64 threads
  const _Float16* row = Vt + ((size_t)bf * CN + c) * HWG;
  float s = 0.f;
  for (int q = 0; q < HWG; ++q) s += (float)row[q];
  gm[bf * CN + c] = s * (1.0f / HWG);
}

// ---------------- cosine-sim top-2 group remap (+identity for ia) ----------------
__global__ void k_sim(const float* __restrict__ gm, int* __restrict__ qia,
                      int* __restrict__ qca) {
  int j = threadIdx.x;
  if (j >= BN * FACTOR) return;
  int b = j / FACTOR, jg = j % FACTOR;
  const float* gb = gm + b * FACTOR * CN;
  float nj = 0.f;
  for (int c = 0; c < CN; ++c) { float v = gb[jg * CN + c]; nj = fmaf(v, v, nj); }
  nj = sqrtf(nj);
  float b1 = -1e30f, b2 = -1e30f;
  int i1 = 0, i2 = 0;
  for (int i = 0; i < FACTOR; ++i) {
    float dot = 0.f, ni = 0.f;
    for (int c = 0; c < CN; ++c) {
      float a = gb[i * CN + c], qv = gb[jg * CN + c];
      dot = fmaf(a, qv, dot);
      ni  = fmaf(a, a, ni);
    }
    ni = sqrtf(ni);
    float cosv = dot / fmaxf(ni * nj, 1e-8f);
    if (cosv > b1)      { b2 = b1; i2 = i1; b1 = cosv; i1 = i; }
    else if (cosv > b2) { b2 = cosv; i2 = i; }
  }
  qca[j] = i2;   // second most similar group (top2[...,1])
  qia[j] = jg;   // identity for intra attention
}

// ---------------- flash-style group attention with softmax_1 ----------------
// one wave per (bf, 16-row q block); logits are cosines in [-1,1] -> no max-sub needed.
__global__ void __launch_bounds__(128, 2)
k_attn(const _Float16* __restrict__ Qn, const _Float16* __restrict__ Kn,
       const _Float16* __restrict__ Vt, const int* __restrict__ qgrp,
       float* __restrict__ Out) {
  __shared__ _Float16 ldsP[4][16 * 32];   // wave-private P tile (C-layout -> A-layout bounce)
  const int lane = threadIdx.x & 31;
  const int wib  = threadIdx.x >> 5;
  const int unit = blockIdx.x * 4 + wib;  // 2048 units total
  const int bf   = unit >> 7;             // [0,16)
  const int qblk = unit & 127;            // [0,128)
  const int b    = bf >> 3;
  const int q0   = qblk << 4;
  const int qg   = qgrp[bf];              // Q source group (identity or sim remap)

  const _Float16* Qbase = Qn + ((size_t)(b * FACTOR + qg) * HWG + q0) * CN;
  const _Float16* Kbase = Kn + (size_t)bf * HWG * CN;
  const _Float16* Vbase = Vt + (size_t)bf * CN * HWG;
  float*          Obase = Out + ((size_t)bf * HWG + q0) * CN;

  const int  rlo   = lane & 15;
  const bool hihlf = lane >= 16;
  const int  lowK  = hihlf ? 8 : 0;    // unused; A-frags use khalf too (K 0..15 vs 16..31)
  const int  khalf = hihlf ? 16 : 0;   // B-frag: K 0..15 vs 16..31

  // Q A-fragments (16x32 f16), two channel chunks, kept in registers all iterations
  const half16 qa0 = ld_a2(Qbase + rlo * CN + khalf,      Qbase + rlo * CN + khalf + 8);
  const half16 qa1 = ld_a2(Qbase + rlo * CN + 32 + khalf, Qbase + rlo * CN + 40 + khalf);

  float8 o0 = {}, o1 = {}, o2 = {}, o3 = {};   // O (16 x 64) f32 accumulators
  float dacc[8];
#pragma unroll
  for (int r = 0; r < 8; ++r) dacc[r] = 0.f;
  _Float16* P = ldsP[wib];

  for (int k0 = 0; k0 < HWG; k0 += 32) {
    // ---- S = Q K^T for a 16x32 key chunk, as two 16x16 tiles ----
#pragma unroll
    for (int t = 0; t < 2; ++t) {
      const int kb = k0 + t * 16 + rlo;  // key index for this lane column
      const half16 kf0 = ld_a2(Kbase + (size_t)kb * CN + khalf,
                               Kbase + (size_t)kb * CN + khalf + 8);
      const half16 kf1 = ld_a2(Kbase + (size_t)kb * CN + 32 + khalf,
                               Kbase + (size_t)kb * CN + 40 + khalf);
      float8 s = {};
      s = wmma_f32_16x16x32_f16(qa0, kf0, s);
      s = wmma_f32_16x16x32_f16(qa1, kf1, s);
      // exp (softmax_1 numerator), denom accumulate, stage P tile in LDS
#pragma unroll
      for (int r = 0; r < 8; ++r) {
        float e = __expf(s[r]);
        dacc[r] += e;
        P[(r + (hihlf ? 8 : 0)) * 32 + t * 16 + rlo] = (_Float16)e;
      }
    }
    asm volatile("s_wait_dscnt 0" ::: "memory");
    // reload P as A-fragment (16x32)
    half16 pa;
#pragma unroll
    for (int i = 0; i < 8; ++i) {
      pa[i]     = P[rlo * 32 + khalf + i];
      pa[8 + i] = P[rlo * 32 + khalf + 8 + i];
    }
    asm volatile("" ::: "memory");
    // ---- O += P V : four 16-channel tiles; Vt is channel-major so B-frags are contiguous
    const _Float16* vcol = Vbase + (size_t)rlo * HWG + k0 + khalf;
    o0 = wmma_f32_16x16x32_f16(pa, ld_a2(vcol,                 vcol + 8),             o0);
    o1 = wmma_f32_16x16x32_f16(pa, ld_a2(vcol + 16 * HWG,      vcol + 16 * HWG + 8),  o1);
    o2 = wmma_f32_16x16x32_f16(pa, ld_a2(vcol + 32 * HWG,      vcol + 32 * HWG + 8),  o2);
    o3 = wmma_f32_16x16x32_f16(pa, ld_a2(vcol + 48 * HWG,      vcol + 48 * HWG + 8),  o3);
  }

  // denom: reduce over the 16 lanes of each half-wave (columns of the C-layout tiles)
#pragma unroll
  for (int r = 0; r < 8; ++r) {
    float v = dacc[r];
    v += __shfl_xor(v, 1, 16);
    v += __shfl_xor(v, 2, 16);
    v += __shfl_xor(v, 4, 16);
    v += __shfl_xor(v, 8, 16);
    dacc[r] = 1.0f / (v + 1.0f);   // softmax_1: sum(exp) + 1
  }
#pragma unroll
  for (int r = 0; r < 8; ++r) {
    float* orow = Obase + (size_t)(r + (hihlf ? 8 : 0)) * CN + rlo;
    orow[0]  = o0[r] * dacc[r];
    orow[16] = o1[r] * dacc[r];
    orow[32] = o2[r] * dacc[r];
    orow[48] = o3[r] * dacc[r];
  }
}

// ---------------- combine ia/ca, scatter back to original order ----------------
__global__ void k_combine(const float* __restrict__ ia, const float* __restrict__ ca,
                          const float* __restrict__ sa, const float* __restrict__ cap,
                          const int* __restrict__ perm, float* __restrict__ comb) {
  int t = blockIdx.x * blockDim.x + threadIdx.x;
  if (t >= BN * HWN) return;
  int p = t % HWN, b = t / HWN;
  int g = p >> 11, q = p & 2047;
  int orig = perm[b * HWN + p];
  size_t srow = ((size_t)(b * FACTOR + g) * HWG + q) * CN;
  for (int c = 0; c < CN; ++c) {
    float v = ia[srow + c] * sa[c] + ca[srow + c] * cap[c];
    comb[(size_t)(b * CN + c) * HWN + orig] = v;   // scatter == gather by inverse perm
  }
}

// ---------------- 1x1 projection ----------------
__global__ void k_proj(const float* __restrict__ in, const float* __restrict__ wp,
                       float* __restrict__ out) {
  int t = blockIdx.x * blockDim.x + threadIdx.x;
  if (t >= BN * CN * HWN) return;
  int p  = t % HWN;
  int oc = (t / HWN) % CN;
  int b  = t / (HWN * CN);
  const float* xb   = in + (size_t)b * CN * HWN + p;
  const float* wrow = wp + oc * CN;
  float s = 0.f;
#pragma unroll 8
  for (int ic = 0; ic < CN; ++ic) s = fmaf(wrow[ic], xb[(size_t)ic * HWN], s);
  out[t] = s;
}

// ---------------- launch ----------------
extern "C" void kernel_launch(void* const* d_in, const int* in_sizes, int n_in,
                              void* d_out, int out_size, void* d_ws, size_t ws_size,
                              hipStream_t stream) {
  (void)in_sizes; (void)n_in; (void)out_size; (void)ws_size;
  const float* feats = (const float*)d_in[0];
  const float* idx   = (const float*)d_in[1];
  const float* wqkv  = (const float*)d_in[2];
  const float* wdw   = (const float*)d_in[3];
  const float* wproj = (const float*)d_in[4];
  const float* sa    = (const float*)d_in[5];
  const float* cap   = (const float*)d_in[6];

  char* ws = (char*)d_ws;
  size_t off = 0;
  auto alloc = [&](size_t bytes) -> void* {
    void* p = ws + off;
    off = (off + bytes + 255) & ~(size_t)255;
    return p;
  };
  float*    bufA = (float*)alloc((size_t)BN * QKVC * HWN * 4);   // qkv1; later ia/ca/comb
  float*    bufB = (float*)alloc((size_t)BN * QKVC * HWN * 4);   // qkv2
  float*    skey = (float*)alloc((size_t)BN * HWN * 4);
  int*      sval = (int*)  alloc((size_t)BN * HWN * 4);
  _Float16* Qn1  = (_Float16*)alloc((size_t)BN * FACTOR * HWG * CN * 2);
  _Float16* Kn1  = (_Float16*)alloc((size_t)BN * FACTOR * HWG * CN * 2);
  _Float16* Qn2  = (_Float16*)alloc((size_t)BN * FACTOR * HWG * CN * 2);
  _Float16* Kn2  = (_Float16*)alloc((size_t)BN * FACTOR * HWG * CN * 2);
  _Float16* Vt   = (_Float16*)alloc((size_t)BN * FACTOR * HWG * CN * 2);
  float*    gm   = (float*)alloc((size_t)BN * FACTOR * CN * 4);
  int*      qia  = (int*)alloc(64);
  int*      qca  = (int*)alloc(64);
  // reuse dead qkv1 buffer (bufA is free after k_dw):
  const size_t atnN = (size_t)BN * FACTOR * HWG * CN;  // 2,097,152 floats
  float* iaOut = bufA;
  float* caOut = bufA + atnN;
  float* comb  = bufA + 2 * atnN;

  const int TB = 256;
  const int nQKV = BN * QKVC * HWN;
  const int nBH  = BN * HWN;
  const int nOut = BN * CN * HWN;

  k_qkv_pw<<<(nQKV + TB - 1) / TB, TB, 0, stream>>>(feats, idx, wqkv, bufA);
  k_dw<<<(nQKV + TB - 1) / TB, TB, 0, stream>>>(bufA, wdw, bufB);

  k_sort_init<<<(nBH + TB - 1) / TB, TB, 0, stream>>>(idx, skey, sval);
  for (int k = 2; k <= HWN; k <<= 1)
    for (int j = k >> 1; j > 0; j >>= 1)
      k_bitonic<<<(nBH + TB - 1) / TB, TB, 0, stream>>>(skey, sval, k, j);

  k_gather<<<(nBH + TB - 1) / TB, TB, 0, stream>>>(bufB, sval, Qn1, Kn1, Qn2, Kn2, Vt);
  k_gmean<<<BN * FACTOR, CN, 0, stream>>>(Vt, gm);
  k_sim<<<1, 32, 0, stream>>>(gm, qia, qca);

  k_attn<<<512, 128, 0, stream>>>(Qn1, Kn1, Vt, qia, iaOut);   // intra attention
  k_attn<<<512, 128, 0, stream>>>(Qn2, Kn2, Vt, qca, caOut);   // cross attention

  k_combine<<<(nBH + TB - 1) / TB, TB, 0, stream>>>(iaOut, caOut, sa, cap, sval, comb);
  k_proj<<<(nOut + TB - 1) / TB, TB, 0, stream>>>(comb, wproj, (float*)d_out);
}
